// AttnBlock1d_39127152067100
// MI455X (gfx1250) — matrix operand order repaired and checked by reading, not compiled, not run on hardware
//
#include <hip/hip_runtime.h>

// ---------------------------------------------------------------------------
// AttnBlock1d for MI455X (gfx1250, wave32, WMMA 16x16x32 f16 -> f32)
//   B=4, C=512, N=2048, 8 heads x 64 dim, 32 groups GroupNorm
// Pipeline:
//   0) cvt_w  : four 512x512 f32 weights -> f16 (once, L2 resident)
//   1) groupnorm : x[B,C,N] f32 -> Ht[B,N,C] f16 (N-major)
//   2) gemm_nt q : qt[B,N,C] = (Ht * q_w^T + q_b) * 512^-0.5  (scale folded)
//      gemm_nt k : kt[B,N,C] =  Ht * k_w^T + k_b
//      gemm_nt v : v [B,C,N] =  v_w * Ht^T + v_b
//   3) flash_attn : online softmax; K/V tiles staged in LDS via
//      global_load_async_to_lds_b128 (ASYNCcnt), double-buffered
//   4) gemm_nt proj : out[B,C,N] f32 = proj_w * attnT^T + proj_b + x
// ---------------------------------------------------------------------------

typedef __attribute__((ext_vector_type(8)))  _Float16 v8h;
typedef __attribute__((ext_vector_type(16))) _Float16 v16h;
typedef __attribute__((ext_vector_type(8)))  float    v8f;
typedef __attribute__((ext_vector_type(4)))  float    v4f;

#define BB 4
#define CC 512
#define NN 2048
#define NH 8
#define HC 64
#define NG 32

// ---- WMMA wrapper ----------------------------------------------------------
__device__ __forceinline__ v8f wmma_f32_f16(v16h a, v16h b, v8f c) {
  return __builtin_amdgcn_wmma_f32_16x16x32_f16(
      false, a, false, b, (short)0, c, false, false);
}

// ---- Fragment loaders (CDNA5 16x16x32 f16 layouts, wave32) -----------------
// A (16x32 MxK): lane l -> row (l&15), K = 8*(l>>4)+{0..7} and 16+8*(l>>4)+{0..7}
__device__ __forceinline__ v16h frag_a(const _Float16* p, int ld, int lane) {
  int m = lane & 15, kb = (lane >> 4) * 8;
  const _Float16* q = p + (size_t)m * ld + kb;
  v8h lo = *(const v8h*)q;
  v8h hi = *(const v8h*)(q + 16);
  return __builtin_shufflevector(lo, hi, 0,1,2,3,4,5,6,7,8,9,10,11,12,13,14,15);
}
// B (32x16 KxN) from row-major NxK source: lane l -> col (l&15), K = 8*(l>>4)+{0..7} and 16+8*(l>>4)+{0..7}
__device__ __forceinline__ v16h frag_b(const _Float16* p, int ld, int lane) {
  int n = lane & 15, kb = (lane >> 4) * 8;
  const _Float16* q = p + (size_t)n * ld + kb;
  v8h lo = *(const v8h*)q;
  v8h hi = *(const v8h*)(q + 16);
  return __builtin_shufflevector(lo, hi, 0,1,2,3,4,5,6,7,8,9,10,11,12,13,14,15);
}

// ---- gfx1250 async copy: 16B global -> LDS, tracked by ASYNCcnt ------------
__device__ __forceinline__ void async_cp16(const _Float16* __restrict__ g,
                                           const _Float16* l) {
  // LDS byte address = low 32 bits of the generic shared pointer
  unsigned la = (unsigned)(unsigned long long)(const void*)l;
  asm volatile("global_load_async_to_lds_b128 %0, %1, off"
               :: "v"(la), "v"(g) : "memory");
}
__device__ __forceinline__ void wait_async0() {
#if __has_builtin(__builtin_amdgcn_s_wait_asynccnt)
  __builtin_amdgcn_s_wait_asynccnt(0);
#else
  asm volatile("s_wait_asynccnt 0x0" ::: "memory");
#endif
}

// ---- 0) f32 -> f16 weight conversion (8 elems / thread) --------------------
__global__ __launch_bounds__(256)
void cvt_w_kernel(const float* __restrict__ src, _Float16* __restrict__ dst) {
  size_t i = ((size_t)blockIdx.x * 256 + threadIdx.x) * 8;
  v4f a = *(const v4f*)(src + i);
  v4f b = *(const v4f*)(src + i + 4);
  v8h o;
#pragma unroll
  for (int j = 0; j < 4; ++j) { o[j] = (_Float16)a[j]; o[j + 4] = (_Float16)b[j]; }
  *(v8h*)(dst + i) = o;
}

// ---- 1) GroupNorm, f32 [B,C,N] -> f16 N-major Ht[B,N,C] --------------------
__global__ __launch_bounds__(256)
void groupnorm_kernel(const float* __restrict__ x,
                      const float* __restrict__ gamma,
                      const float* __restrict__ beta,
                      _Float16* __restrict__ ht) {
  const int b = blockIdx.x >> 5;
  const int g = blockIdx.x & 31;
  const int tid = threadIdx.x;
  const int cnt = (CC / NG) * NN;    // 32768
  const float* xb = x + ((size_t)b * CC + (size_t)g * (CC / NG)) * NN;

  float s = 0.f, s2 = 0.f;
  for (int i = tid; i < cnt; i += 256) { float v = xb[i]; s += v; s2 += v * v; }

  __shared__ float sh1[256], sh2[256];
  sh1[tid] = s; sh2[tid] = s2;
  __syncthreads();
  for (int off = 128; off > 0; off >>= 1) {
    if (tid < off) { sh1[tid] += sh1[tid + off]; sh2[tid] += sh2[tid + off]; }
    __syncthreads();
  }
  const float inv  = 1.0f / (float)cnt;
  const float mean = sh1[0] * inv;
  const float var  = sh2[0] * inv - mean * mean;
  const float rstd = rsqrtf(var + 1e-6f);

  for (int i = tid; i < cnt; i += 256) {
    int c = i >> 11, n = i & (NN - 1);
    int cg = g * (CC / NG) + c;
    float val = (xb[i] - mean) * rstd * gamma[cg] + beta[cg];
    ht[((size_t)b * NN + n) * CC + cg] = (_Float16)val;
  }
}

// ---- 2/4) NT GEMM: D[M,N] = (A[M,K] * B[N,K]^T + bias) * scale (+resid) ----
// 256 threads = 8 waves, wave grid 4(M) x 2(N); wave tile 32x64; block 128x128.
template <bool BIAS_ROW, bool OUT_RESID>
__global__ __launch_bounds__(256)
void gemm_nt_kernel(const _Float16* __restrict__ Ab, const _Float16* __restrict__ Bb,
                    const float* __restrict__ bias,
                    const float* __restrict__ resid,
                    void* __restrict__ outv,
                    int M, int N, int K, float outScale,
                    size_t aStride, size_t bStride, size_t oStride, size_t rStride) {
  const int tid = threadIdx.x, lane = tid & 31, w = tid >> 5;
  const int mW = blockIdx.x * 128 + (w & 3) * 32;
  const int nW = blockIdx.y * 128 + (w >> 2) * 64;
  const int bz = blockIdx.z;
  const _Float16* A = Ab + aStride * bz;
  const _Float16* B = Bb + bStride * bz;

  v8f acc[2][4] = {};
  for (int k0 = 0; k0 < K; k0 += 32) {
    v16h a0 = frag_a(A + (size_t)mW * K + k0, K, lane);
    v16h a1 = frag_a(A + (size_t)(mW + 16) * K + k0, K, lane);
#pragma unroll
    for (int t = 0; t < 4; ++t) {
      v16h bf = frag_b(B + (size_t)(nW + t * 16) * K + k0, K, lane);
      acc[0][t] = wmma_f32_f16(a0, bf, acc[0][t]);
      acc[1][t] = wmma_f32_f16(a1, bf, acc[1][t]);
    }
  }

  const int half = lane >> 4, nl = lane & 15;
#pragma unroll
  for (int u = 0; u < 2; ++u) {
#pragma unroll
    for (int t = 0; t < 4; ++t) {
#pragma unroll
      for (int r = 0; r < 8; ++r) {
        int row = mW + u * 16 + r + 8 * half;
        int col = nW + t * 16 + nl;
        float v = (acc[u][t][r] + (BIAS_ROW ? bias[row] : bias[col])) * outScale;
        size_t oidx = oStride * bz + (size_t)row * N + col;
        if constexpr (OUT_RESID) {
          ((float*)outv)[oidx] = v + resid[rStride * bz + (size_t)row * N + col];
        } else {
          ((_Float16*)outv)[oidx] = (_Float16)v;
        }
      }
    }
  }
}

// ---- 3) Flash attention per (qblock, head, batch) --------------------------
// 128 threads = 4 waves; wave owns 16 query rows; 64-key tiles.
// K/V tiles staged once per block into LDS with async copies (double-buffered);
// copy of tile i+1 overlaps compute of tile i. Q pre-scaled by 512^-0.5.
__global__ __launch_bounds__(128)
void flash_attn_kernel(const _Float16* __restrict__ qt,   // [B,N,C] (scaled)
                       const _Float16* __restrict__ kt,   // [B,N,C]
                       const _Float16* __restrict__ vm,   // [B,C,N]
                       _Float16* __restrict__ attnT) {    // [B,N,C]
  const int tid = threadIdx.x, lane = tid & 31, w = tid >> 5;
  const int h = blockIdx.y, b = blockIdx.z;
  const int qRow = blockIdx.x * 64 + w * 16;

  const _Float16* qB = qt + (size_t)b * NN * CC;
  const _Float16* kB = kt + (size_t)b * NN * CC + (size_t)h * HC;
  const _Float16* vB = vm + (size_t)b * CC * NN + (size_t)h * HC * NN;

  __shared__ __align__(32) _Float16 Kbuf[2][64][64];   // [key][dim]   16 KB
  __shared__ __align__(32) _Float16 Vbuf[2][64][64];   // [chan][key]  16 KB
  __shared__ __align__(32) _Float16 Plds[4][16][64];   // wave-private  8 KB

  v16h aq0 = frag_a(qB + (size_t)qRow * CC + h * HC + 0, CC, lane);
  v16h aq1 = frag_a(qB + (size_t)qRow * CC + h * HC + 32, CC, lane);

  v8f o[4] = {};
  float mrow[8], lrow[8];
#pragma unroll
  for (int r = 0; r < 8; ++r) { mrow[r] = -1e30f; lrow[r] = 0.f; }
  const int half = lane >> 4, nl = lane & 15;

  // issue async copies for tile at key offset kb into buffer `bf`
  auto issue_tile = [&](int bf, int kb) {
#pragma unroll
    for (int j = 0; j < 4; ++j) {
      int chunk = tid + j * 128;          // 512 x 16B chunks per tile
      int row = chunk >> 3;               // 64 rows
      int c8  = (chunk & 7) * 8;          // 8 halves per chunk
      async_cp16(kB + (size_t)(kb + row) * CC + c8, &Kbuf[bf][row][c8]);
      async_cp16(vB + (size_t)row * NN + kb + c8,   &Vbuf[bf][row][c8]);
    }
  };

  issue_tile(0, 0);
  for (int it = 0; it < NN / 64; ++it) {
    const int bf = it & 1;
    wait_async0();        // this wave's copies of tile `it` landed
    __syncthreads();      // all waves landed + all reads of buffer bf^1 retired
    if (it + 1 < NN / 64) issue_tile(bf ^ 1, (it + 1) * 64);

    // S[16q x 64k] = Qs * K^T  (operands now ds_load_b128 from LDS)
    v8f s[4] = {};
#pragma unroll
    for (int t = 0; t < 4; ++t) {
      v16h bk0 = frag_b(&Kbuf[bf][t * 16][0] + 0, 64, lane);
      s[t] = wmma_f32_f16(aq0, bk0, s[t]);
      v16h bk1 = frag_b(&Kbuf[bf][t * 16][0] + 32, 64, lane);
      s[t] = wmma_f32_f16(aq1, bk1, s[t]);
    }
    // online softmax; rows at (r + 8*half), reductions within 16-lane halves
#pragma unroll
    for (int r = 0; r < 8; ++r) {
      float mx = fmaxf(fmaxf(s[0][r], s[1][r]), fmaxf(s[2][r], s[3][r]));
#pragma unroll
      for (int msk = 1; msk < 16; msk <<= 1) mx = fmaxf(mx, __shfl_xor(mx, msk, 32));
      float mnew = fmaxf(mrow[r], mx);
      float alpha = __expf(mrow[r] - mnew);
      mrow[r] = mnew;
      float rs = 0.f;
#pragma unroll
      for (int t = 0; t < 4; ++t) {
        float e = __expf(s[t][r] - mnew);
        s[t][r] = e;
        rs += e;
      }
#pragma unroll
      for (int msk = 1; msk < 16; msk <<= 1) rs += __shfl_xor(rs, msk, 32);
      lrow[r] = lrow[r] * alpha + rs;
#pragma unroll
      for (int t = 0; t < 4; ++t) o[t][r] *= alpha;
    }
    // P -> LDS (accumulator layout -> A-frag layout round trip, wave-private)
#pragma unroll
    for (int t = 0; t < 4; ++t)
#pragma unroll
      for (int r = 0; r < 8; ++r)
        Plds[w][r + 8 * half][t * 16 + nl] = (_Float16)s[t][r];

    v16h ap0 = frag_a(&Plds[w][0][0], 64, lane);
    v16h ap1 = frag_a(&Plds[w][0][0] + 32, 64, lane);
    // O[16q x 64c] += P * V^T   (Vbuf is [chan][key] -> B-frag contiguous)
#pragma unroll
    for (int t = 0; t < 4; ++t) {
      v16h bv0 = frag_b(&Vbuf[bf][t * 16][0] + 0, 64, lane);
      o[t] = wmma_f32_f16(ap0, bv0, o[t]);
      v16h bv1 = frag_b(&Vbuf[bf][t * 16][0] + 32, 64, lane);
      o[t] = wmma_f32_f16(ap1, bv1, o[t]);
    }
  }

#pragma unroll
  for (int r = 0; r < 8; ++r) {
    float invl = 1.0f / lrow[r];
    int q = qRow + r + 8 * half;
#pragma unroll
    for (int t = 0; t < 4; ++t) {
      int c = h * HC + t * 16 + nl;
      attnT[((size_t)b * NN + q) * CC + c] = (_Float16)(o[t][r] * invl);
    }
  }
}

// ---------------------------------------------------------------------------
extern "C" void kernel_launch(void* const* d_in, const int* in_sizes, int n_in,
                              void* d_out, int out_size, void* d_ws, size_t ws_size,
                              hipStream_t stream) {
  (void)in_sizes; (void)n_in; (void)out_size; (void)ws_size;
  const float* x      = (const float*)d_in[0];
  const float* gamma  = (const float*)d_in[1];
  const float* beta   = (const float*)d_in[2];
  const float* q_w    = (const float*)d_in[3];
  const float* q_b    = (const float*)d_in[4];
  const float* k_w    = (const float*)d_in[5];
  const float* k_b    = (const float*)d_in[6];
  const float* v_w    = (const float*)d_in[7];
  const float* v_b    = (const float*)d_in[8];
  const float* proj_w = (const float*)d_in[9];
  const float* proj_b = (const float*)d_in[10];
  float* out = (float*)d_out;

  const size_t PLANE = (size_t)BB * NN * CC * sizeof(_Float16);   // 8 MB
  const size_t WSZ   = (size_t)CC * CC;                           // 262144 elems
  _Float16* Ht    = (_Float16*)d_ws;
  _Float16* qt    = (_Float16*)((char*)d_ws + 1 * PLANE);
  _Float16* ktb   = (_Float16*)((char*)d_ws + 2 * PLANE);
  _Float16* vmb   = (_Float16*)((char*)d_ws + 3 * PLANE);
  _Float16* attnT = (_Float16*)((char*)d_ws + 4 * PLANE);
  _Float16* qwh   = (_Float16*)((char*)d_ws + 5 * PLANE);
  _Float16* kwh   = qwh + 1 * WSZ;
  _Float16* vwh   = qwh + 2 * WSZ;
  _Float16* pwh   = qwh + 3 * WSZ;

  const size_t actS = (size_t)NN * CC;
  const size_t zero = 0;
  const int cvtGrid = (int)(WSZ / (256 * 8));   // 128 blocks

  // 0) weights -> f16 (once per launch; tiny, L2 resident afterwards)
  cvt_w_kernel<<<cvtGrid, 256, 0, stream>>>(q_w, qwh);
  cvt_w_kernel<<<cvtGrid, 256, 0, stream>>>(k_w, kwh);
  cvt_w_kernel<<<cvtGrid, 256, 0, stream>>>(v_w, vwh);
  cvt_w_kernel<<<cvtGrid, 256, 0, stream>>>(proj_w, pwh);

  // 1) GroupNorm -> Ht[B,N,C] f16
  groupnorm_kernel<<<dim3(BB * NG), 256, 0, stream>>>(x, gamma, beta, Ht);

  // 2) q (scale folded), k : [N,C];  v : [C,N]
  const float qscale = 0.044194173824159216f;   // 512^-0.5
  gemm_nt_kernel<false, false><<<dim3(NN / 128, CC / 128, BB), 256, 0, stream>>>(
      Ht, qwh, q_b, nullptr, qt, NN, CC, CC, qscale, actS, zero, actS, zero);
  gemm_nt_kernel<false, false><<<dim3(NN / 128, CC / 128, BB), 256, 0, stream>>>(
      Ht, kwh, k_b, nullptr, ktb, NN, CC, CC, 1.0f, actS, zero, actS, zero);
  gemm_nt_kernel<true, false><<<dim3(CC / 128, NN / 128, BB), 256, 0, stream>>>(
      vwh, Ht, v_b, nullptr, vmb, CC, NN, CC, 1.0f, zero, actS, actS, zero);

  // 3) flash attention (async LDS staging)
  flash_attn_kernel<<<dim3(NN / 64, NH, BB), 128, 0, stream>>>(qt, ktb, vmb, attnT);

  // 4) proj + bias + residual -> f32 out[B,C,N]
  gemm_nt_kernel<true, true><<<dim3(CC / 128, NN / 128, BB), 256, 0, stream>>>(
      pwh, attnT, proj_b, x, out, CC, NN, CC, 1.0f, zero, actS, actS, actS);
}
